// FalconAttention_59992103190828
// MI455X (gfx1250) — matrix operand order repaired and checked by reading, not compiled, not run on hardware
//
#include <hip/hip_runtime.h>
#include <hip/hip_bf16.h>

#define S_LEN   2048
#define HID     4096
#define NH      32
#define NKV     8
#define HD      128
#define QKV_OUT 6144   // (8*2+32)*128

typedef __attribute__((ext_vector_type(16))) __bf16       v16bf;
typedef __attribute__((ext_vector_type(8)))  float        v8f;
typedef __attribute__((ext_vector_type(4)))  unsigned int uint4v;
typedef int vint4 __attribute__((__vector_size__(16)));

#if __has_builtin(__builtin_amdgcn_global_load_async_to_lds_b128) && \
    __has_builtin(__builtin_amdgcn_s_wait_asynccnt)
#define HAVE_ASYNC_LDS 1
#else
#define HAVE_ASYNC_LDS 0
#endif

__device__ __forceinline__ unsigned short f2bf(float x) {
  unsigned int u = __float_as_uint(x);
  unsigned int r = u + 0x7FFFu + ((u >> 16) & 1u);   // round-to-nearest-even
  return (unsigned short)(r >> 16);
}

// 16-byte global -> LDS copy. CDNA5 path: ASYNCcnt-tracked DMA, no VGPR staging.
__device__ __forceinline__ void async_copy16(const unsigned short* g, unsigned short* l) {
#if HAVE_ASYNC_LDS
  __builtin_amdgcn_global_load_async_to_lds_b128(
      (__attribute__((address_space(1))) vint4*)(g),
      (__attribute__((address_space(3))) vint4*)(l), 0, 0);
#else
  *(uint4v*)l = *(const uint4v*)g;
#endif
}

// A-matrix fragment (16x32 bf16): lane<16 holds K{0..7,16..23},
// lane>=16 holds K{8..15,24..31}; caller passes p = row_base + (lane&16 ? 8 : 0)
__device__ __forceinline__ v16bf load_fragA(const unsigned short* p) {
  union { uint4v u[2]; v16bf v; } un;
  un.u[0] = *(const uint4v*)(p);
  un.u[1] = *(const uint4v*)(p + 16);
  return un.v;
}

// B-matrix fragment (32x16 bf16): lane = column, same K order as A-fragment;
// caller passes p = col_base + (lane&16 ? 8 : 0)
__device__ __forceinline__ v16bf load_fragB(const unsigned short* p) {
  union { uint4v u[2]; v16bf v; } un;
  un.u[0] = *(const uint4v*)(p);
  un.u[1] = *(const uint4v*)(p + 16);
  return un.v;
}

// ---------------------------------------------------------------------------
// fp32 -> bf16 bulk convert (grid-stride)
// ---------------------------------------------------------------------------
__global__ __launch_bounds__(256) void f32_to_bf16_kernel(
    const float* __restrict__ src, unsigned short* __restrict__ dst, size_t n) {
  size_t i = (size_t)blockIdx.x * blockDim.x + threadIdx.x;
  size_t stride = (size_t)gridDim.x * blockDim.x;
  for (; i < n; i += stride) dst[i] = f2bf(src[i]);
}

// ---------------------------------------------------------------------------
// C[M,N] f32 = A[M,K] bf16 (row-major) * B[N,K] bf16 (row-major, i.e. B^T)
// Block = 256 threads (8 waves), tile 128x128, K-step 32.
// Double-buffered LDS filled by GLOBAL_LOAD_ASYNC_TO_LDS_B128 (ASYNCcnt),
// so the next K-tile streams in while WMMAs consume the current one.
// Wave grid 4(M) x 2(N); wave tile 32x64 -> 2x4 WMMA accumulators.
// ---------------------------------------------------------------------------
__global__ __launch_bounds__(256) void gemm_bf16_kernel(
    const unsigned short* __restrict__ A, const unsigned short* __restrict__ B,
    float* __restrict__ C, int M, int N, int K, int lda, int ldb, int ldc) {
  __shared__ unsigned short As[2][128][40];   // +8 pad keeps 16B alignment
  __shared__ unsigned short Bs[2][128][40];

  const int tid    = threadIdx.x;
  const int lane   = tid & 31;
  const int wave   = tid >> 5;
  const int wm     = wave & 3;        // 0..3  (M)
  const int wn     = wave >> 2;       // 0..1  (N)
  const int lanelo = lane & 15;
  const int half   = lane >> 4;
  const int kbA    = half ? 8 : 0;
  const int kbB    = half ? 8 : 0;
  const int m0     = blockIdx.y * 128;
  const int n0     = blockIdx.x * 128;

  // this thread's two 16B chunks of each 128x32 tile (512 chunks / 256 thr)
  const int rowc0 = tid >> 2;
  const int rowc1 = (tid + 256) >> 2;
  const int colc  = (tid & 3) * 8;

  v8f acc[2][4];
#pragma unroll
  for (int mt = 0; mt < 2; ++mt)
#pragma unroll
    for (int nt = 0; nt < 4; ++nt)
      acc[mt][nt] = (v8f){0.f,0.f,0.f,0.f,0.f,0.f,0.f,0.f};

  // prime buffer 0 with the first K-tile (4 async DMAs per thread)
  async_copy16(A + (size_t)(m0 + rowc0) * lda + colc, &As[0][rowc0][colc]);
  async_copy16(A + (size_t)(m0 + rowc1) * lda + colc, &As[0][rowc1][colc]);
  async_copy16(B + (size_t)(n0 + rowc0) * ldb + colc, &Bs[0][rowc0][colc]);
  async_copy16(B + (size_t)(n0 + rowc1) * ldb + colc, &Bs[0][rowc1][colc]);

  int cur = 0;
  for (int k0 = 0; k0 < K; k0 += 32) {
    const int nxt = cur ^ 1;
    const bool more = (k0 + 32) < K;
    if (more) {   // stream next tile into the free buffer
      const int k1 = k0 + 32;
      async_copy16(A + (size_t)(m0 + rowc0) * lda + k1 + colc, &As[nxt][rowc0][colc]);
      async_copy16(A + (size_t)(m0 + rowc1) * lda + k1 + colc, &As[nxt][rowc1][colc]);
      async_copy16(B + (size_t)(n0 + rowc0) * ldb + k1 + colc, &Bs[nxt][rowc0][colc]);
      async_copy16(B + (size_t)(n0 + rowc1) * ldb + k1 + colc, &Bs[nxt][rowc1][colc]);
    }
#if HAVE_ASYNC_LDS
    // retire the current tile's 4 DMAs; leave the 4 next-tile DMAs in flight
    if (more) __builtin_amdgcn_s_wait_asynccnt(4);
    else      __builtin_amdgcn_s_wait_asynccnt(0);
#endif
    __syncthreads();

    v16bf af[2], bf[4];
#pragma unroll
    for (int mt = 0; mt < 2; ++mt)
      af[mt] = load_fragA(&As[cur][wm * 32 + mt * 16 + lanelo][kbA]);
#pragma unroll
    for (int nt = 0; nt < 4; ++nt)
      bf[nt] = load_fragB(&Bs[cur][wn * 64 + nt * 16 + lanelo][kbB]);
#pragma unroll
    for (int mt = 0; mt < 2; ++mt)
#pragma unroll
      for (int nt = 0; nt < 4; ++nt)
        acc[mt][nt] = __builtin_amdgcn_wmma_f32_16x16x32_bf16(
            false, af[mt], false, bf[nt], (short)0, acc[mt][nt], false, false);
    __syncthreads();   // buffer `cur` becomes free for iteration k0+64
    cur = nxt;
  }

  // C layout: element (v, lane) -> M = v + 8*half, N = lanelo
#pragma unroll
  for (int mt = 0; mt < 2; ++mt)
#pragma unroll
    for (int nt = 0; nt < 4; ++nt)
#pragma unroll
      for (int v = 0; v < 8; ++v) {
        int m = m0 + wm * 32 + mt * 16 + v + 8 * half;
        int n = n0 + wn * 64 + nt * 16 + lanelo;
        C[(size_t)m * ldc + n] = acc[mt][nt][v];
      }
}

// ---------------------------------------------------------------------------
// RoPE + head rearrange from fused QKV (fp32) into bf16 buffers:
//   q_bf [NH][S][HD], k_bf [NKV][S][HD], vt_bf [NKV][HD][S] (V transposed)
// ---------------------------------------------------------------------------
#define NQ_WORK ((size_t)S_LEN * NH * 64)
#define NK_WORK ((size_t)S_LEN * NKV * 64)
#define NV_WORK ((size_t)S_LEN * NKV * HD)

__global__ __launch_bounds__(256) void rope_rearrange_kernel(
    const float* __restrict__ qkv, const int* __restrict__ pos_ids,
    unsigned short* __restrict__ q_bf, unsigned short* __restrict__ k_bf,
    unsigned short* __restrict__ vt_bf) {
  size_t tid = (size_t)blockIdx.x * blockDim.x + threadIdx.x;
  if (tid < NQ_WORK) {
    int i = (int)(tid & 63);
    size_t r = tid >> 6;
    int h = (int)(r & (NH - 1));
    int s = (int)(r >> 5);
    int g = h >> 2, j = h & 3;
    const float* src = qkv + (size_t)s * QKV_OUT + (size_t)(g * 6 + j) * HD;
    float x1 = src[i], x2 = src[i + 64];
    float pos = (float)pos_ids[s];
    float invf = __powf(10000.0f, -((float)(2 * i)) / 128.0f);
    float sn, cs; __sincosf(pos * invf, &sn, &cs);
    unsigned short* dst = q_bf + ((size_t)h * S_LEN + s) * HD;
    dst[i]      = f2bf(x1 * cs - x2 * sn);
    dst[i + 64] = f2bf(x2 * cs + x1 * sn);
  } else if (tid < NQ_WORK + NK_WORK) {
    size_t t = tid - NQ_WORK;
    int i = (int)(t & 63);
    size_t r = t >> 6;
    int kv = (int)(r & (NKV - 1));
    int s = (int)(r >> 3);
    const float* src = qkv + (size_t)s * QKV_OUT + (size_t)(kv * 6 + 4) * HD;
    float x1 = src[i], x2 = src[i + 64];
    float pos = (float)pos_ids[s];
    float invf = __powf(10000.0f, -((float)(2 * i)) / 128.0f);
    float sn, cs; __sincosf(pos * invf, &sn, &cs);
    unsigned short* dst = k_bf + ((size_t)kv * S_LEN + s) * HD;
    dst[i]      = f2bf(x1 * cs - x2 * sn);
    dst[i + 64] = f2bf(x2 * cs + x1 * sn);
  } else if (tid < NQ_WORK + NK_WORK + NV_WORK) {
    size_t t = tid - NQ_WORK - NK_WORK;
    int d = (int)(t & (HD - 1));
    size_t r = t >> 7;
    int kv = (int)(r & (NKV - 1));
    int s = (int)(r >> 3);
    float x = qkv[(size_t)s * QKV_OUT + (size_t)(kv * 6 + 5) * HD + d];
    vt_bf[((size_t)kv * HD + d) * S_LEN + s] = f2bf(x);
  }
}

// ---------------------------------------------------------------------------
// Flash attention: 1 wave per block, 16 query rows x 1 head per block.
// Streams keys in tiles of 32; online softmax; all MACs via bf16 WMMA.
// Output: attn_bf [S][NH*HD]  (row-major, ready as GEMM A operand)
// ---------------------------------------------------------------------------
__global__ __launch_bounds__(32) void flash_attn_kernel(
    const unsigned short* __restrict__ Q, const unsigned short* __restrict__ Kc,
    const unsigned short* __restrict__ Vt, unsigned short* __restrict__ O) {
  __shared__ unsigned short Ps[16][40];   // P bounce buffer: C-layout -> A-layout

  const int lane   = threadIdx.x & 31;
  const int lanelo = lane & 15;
  const int half   = lane >> 4;
  const int kbA    = half ? 8 : 0;
  const int kbB    = half ? 8 : 0;
  const int s0     = blockIdx.x * 16;
  const int head   = blockIdx.y;
  const int kv     = head >> 2;

  const unsigned short* qbase = Q  + (size_t)head * S_LEN * HD;
  const unsigned short* kbase = Kc + (size_t)kv   * S_LEN * HD;
  const unsigned short* vbase = Vt + (size_t)kv   * HD * S_LEN;

  // Q tile 16x128 as 4 A-fragments (K = 32 each), kept in registers
  v16bf qf[4];
#pragma unroll
  for (int c = 0; c < 4; ++c)
    qf[c] = load_fragA(qbase + (size_t)(s0 + lanelo) * HD + c * 32 + kbA);

  v8f o[8];
  float mrow[8], lrow[8];
#pragma unroll
  for (int nt = 0; nt < 8; ++nt) o[nt] = (v8f){0.f,0.f,0.f,0.f,0.f,0.f,0.f,0.f};
#pragma unroll
  for (int v = 0; v < 8; ++v) { mrow[v] = -1e30f; lrow[v] = 0.f; }

  const float scale = 0.08838834764831845f;  // 1/sqrt(128)

  for (int kt = 0; kt < S_LEN; kt += 32) {
    if (kt + 32 < S_LEN) {   // CDNA5 global_prefetch_b8 of next K/V tiles
      __builtin_prefetch(kbase + (size_t)(kt + 32 + lane) * HD, 0, 1);
      __builtin_prefetch(vbase + (size_t)(lane * 4) * S_LEN + kt + 32, 0, 1);
    }

    // S = Q * K^T : 16x32 scores as two 16x16 accumulators
    v8f sc[2];
    sc[0] = (v8f){0.f,0.f,0.f,0.f,0.f,0.f,0.f,0.f};
    sc[1] = (v8f){0.f,0.f,0.f,0.f,0.f,0.f,0.f,0.f};
#pragma unroll
    for (int c = 0; c < 4; ++c) {
#pragma unroll
      for (int t = 0; t < 2; ++t) {
        v16bf kf = load_fragB(kbase + (size_t)(kt + t * 16 + lanelo) * HD + c * 32 + kbB);
        sc[t] = __builtin_amdgcn_wmma_f32_16x16x32_bf16(
            false, qf[c], false, kf, (short)0, sc[t], false, false);
      }
    }

    // online softmax; rows live in (v, half) slots, N striped across 16 lanes
    float p[2][8], fac[8];
#pragma unroll
    for (int v = 0; v < 8; ++v) {
      float a0 = sc[0][v] * scale, a1 = sc[1][v] * scale;
      float mv = fmaxf(a0, a1);
      mv = fmaxf(mv, __shfl_xor(mv, 1));
      mv = fmaxf(mv, __shfl_xor(mv, 2));
      mv = fmaxf(mv, __shfl_xor(mv, 4));
      mv = fmaxf(mv, __shfl_xor(mv, 8));
      float mnew = fmaxf(mrow[v], mv);
      float f = __expf(mrow[v] - mnew);
      mrow[v] = mnew; fac[v] = f;
      float p0 = __expf(a0 - mnew), p1 = __expf(a1 - mnew);
      p[0][v] = p0; p[1][v] = p1;
      float rs = p0 + p1;
      rs += __shfl_xor(rs, 1);
      rs += __shfl_xor(rs, 2);
      rs += __shfl_xor(rs, 4);
      rs += __shfl_xor(rs, 8);
      lrow[v] = lrow[v] * f + rs;
    }
#pragma unroll
    for (int nt = 0; nt < 8; ++nt)
#pragma unroll
      for (int v = 0; v < 8; ++v) o[nt][v] *= fac[v];

    // P (C-layout) -> LDS -> A-layout fragment
    __syncthreads();
#pragma unroll
    for (int t = 0; t < 2; ++t)
#pragma unroll
      for (int v = 0; v < 8; ++v)
        Ps[v + 8 * half][t * 16 + lanelo] = f2bf(p[t][v]);
    __syncthreads();
    v16bf pf = load_fragA(&Ps[lanelo][kbA]);

    // O += P * V : V^T stored [d][s] so each B-fragment load is contiguous in keys
#pragma unroll
    for (int nt = 0; nt < 8; ++nt) {
      v16bf vf = load_fragB(vbase + (size_t)(nt * 16 + lanelo) * S_LEN + kt + kbB);
      o[nt] = __builtin_amdgcn_wmma_f32_16x16x32_bf16(
          false, pf, false, vf, (short)0, o[nt], false, false);
    }
  }

  // normalize and emit attn output [s][head*128 + d] in bf16
#pragma unroll
  for (int v = 0; v < 8; ++v) {
    float inv = 1.0f / lrow[v];
    int m = s0 + v + 8 * half;
#pragma unroll
    for (int nt = 0; nt < 8; ++nt)
      O[(size_t)m * (NH * HD) + head * HD + nt * 16 + lanelo] = f2bf(o[nt][v] * inv);
  }
}

// ---------------------------------------------------------------------------
// launch
// ---------------------------------------------------------------------------
extern "C" void kernel_launch(void* const* d_in, const int* in_sizes, int n_in,
                              void* d_out, int out_size, void* d_ws, size_t ws_size,
                              hipStream_t stream) {
  const float* hidden  = (const float*)d_in[0];
  const float* w_qkv   = (const float*)d_in[1];
  const float* w_dense = (const float*)d_in[2];
  const int*   pos_ids = (const int*)d_in[3];

  char* ws = (char*)d_ws;
  unsigned short* h_bf    = (unsigned short*)(ws);                 // 16 MB
  unsigned short* wq_bf   = (unsigned short*)(ws + 16777216ull);   // 48 MB
  unsigned short* wd_bf   = (unsigned short*)(ws + 67108864ull);   // 32 MB
  float*          qkv_f   = (float*)        (ws + 100663296ull);   // 48 MB
  unsigned short* q_bf    = (unsigned short*)(ws + 150994944ull);  // 16 MB
  unsigned short* k_bf    = (unsigned short*)(ws + 167772160ull);  // 4 MB
  unsigned short* vt_bf   = (unsigned short*)(ws + 171966464ull);  // 4 MB
  unsigned short* attn_bf = (unsigned short*)(ws + 176160768ull);  // 16 MB

  f32_to_bf16_kernel<<<4096, 256, 0, stream>>>(hidden,  h_bf,  (size_t)S_LEN * HID);
  f32_to_bf16_kernel<<<8192, 256, 0, stream>>>(w_qkv,   wq_bf, (size_t)QKV_OUT * HID);
  f32_to_bf16_kernel<<<8192, 256, 0, stream>>>(w_dense, wd_bf, (size_t)HID * HID);

  gemm_bf16_kernel<<<dim3(QKV_OUT / 128, S_LEN / 128), 256, 0, stream>>>(
      h_bf, wq_bf, qkv_f, S_LEN, QKV_OUT, HID, HID, HID, QKV_OUT);

  size_t total_work = NQ_WORK + NK_WORK + NV_WORK;   // 7,340,032
  rope_rearrange_kernel<<<(unsigned)((total_work + 255) / 256), 256, 0, stream>>>(
      qkv_f, pos_ids, q_bf, k_bf, vt_bf);

  flash_attn_kernel<<<dim3(S_LEN / 16, NH), 32, 0, stream>>>(q_bf, k_bf, vt_bf, attn_bf);

  gemm_bf16_kernel<<<dim3(HID / 128, S_LEN / 128), 256, 0, stream>>>(
      attn_bf, wd_bf, (float*)d_out, S_LEN, HID, HID, HID, HID, HID);
}
